// MPNN_11252814315887
// MI455X (gfx1250) — matrix operand order repaired and checked by reading, not compiled, not run on hardware
//
#include <hip/hip_runtime.h>
#include <hip/hip_bf16.h>

// ---------------------------------------------------------------------------
// MPNN forward for MI455X (gfx1250, wave32, WMMA).
//
// Refactor: msg(u->v) = relu(A[u] + B[v] + bias)/7 with A = h@W1^T, B = h@W2^T
// => per round ONE [32768,128]x[128,256] GEMM (f16 WMMA, f32 accum) + cheap
// elementwise aggregation. Readout = mean + two [4096,128]x[128,128] GEMMs.
// Working set (h 16MB + AB 32MB) is L2-resident (192MB); compute-bound on WMMA.
// GEMM: 2 M-tiles per wave so each LDS B-fragment feeds 2 independent WMMAs.
// ---------------------------------------------------------------------------

typedef __attribute__((ext_vector_type(16))) _Float16     v16h;
typedef __attribute__((ext_vector_type(8)))  float        v8f;
typedef __attribute__((ext_vector_type(8)))  unsigned int v8u;

#define N_NODES 512          // B*CH
#define T_STEPS 64
#define DIM     128
#define ROWS    (N_NODES * T_STEPS)   // 32768
#define B_SZ    64
#define OUTROWS (B_SZ * T_STEPS)      // 4096

// ------------------------- helper: latents transpose -----------------------
// latents [N][D][T] (f32) -> h [N][T][D]
__global__ __launch_bounds__(256) void transpose_k(const float* __restrict__ lat,
                                                   float* __restrict__ h) {
    size_t idx = (size_t)blockIdx.x * 256 + threadIdx.x;   // over N*T*D
    int d = (int)(idx & (DIM - 1));
    size_t r = idx >> 7;
    int t = (int)(r & (T_STEPS - 1));
    int n = (int)(r >> 6);
    h[idx] = lat[((size_t)n * DIM + d) * T_STEPS + t];
}

// ---------------------- weight pre-conversion kernels -----------------------
// Wcat[r][dp][k] (f16): dp<128 -> W_msg[r][dp][k] ; dp>=128 -> W_msg[r][dp-128][128+k]
__global__ __launch_bounds__(256) void prep_wcat_k(const float* __restrict__ Wmsg,
                                                   _Float16* __restrict__ Wcat) {
    int idx = blockIdx.x * 256 + threadIdx.x;              // over 3*256*128
    int k  = idx & 127;
    int dp = (idx >> 7) & 255;
    int r  = idx >> 15;
    const float* Wr = Wmsg + (size_t)r * DIM * (2 * DIM);
    float w = (dp < DIM) ? Wr[(size_t)dp * 256 + k]
                         : Wr[(size_t)(dp - DIM) * 256 + DIM + k];
    Wcat[idx] = (_Float16)w;
}

__global__ __launch_bounds__(256) void prep_w_k(const float* __restrict__ W,
                                                _Float16* __restrict__ Wh, int n) {
    int idx = blockIdx.x * 256 + threadIdx.x;
    if (idx < n) Wh[idx] = (_Float16)W[idx];
}

// ------------------------------ WMMA GEMM -----------------------------------
// Out[M][Ncols] = X[M][128] * Wh[Ncols][128]^T (+bias, +relu via template).
// Block: 256 threads = 8 waves covering 256 rows; wave w owns 2 M-tiles:
// rows blockIdx*256 + w*16 and +128. Each B fragment feeds 2 independent WMMAs.
// Weight panel staged in 64KB LDS with 16B-chunk XOR swizzle (bank-conflict free).
__device__ __forceinline__ unsigned lds_chunk(unsigned row, unsigned chunk) {
    // row stride = 64 dwords (128 f16); swizzle at 4-dword granularity
    return row * 64u + ((chunk ^ (row & 15u)) << 2);
}

template <bool HAS_BIAS, bool RELU>
__global__ __launch_bounds__(256) void gemm_k(const float* __restrict__ X,
                                              const _Float16* __restrict__ Wh,
                                              float* __restrict__ Out,
                                              const float* __restrict__ bias,
                                              int M, int Ncols) {
    __shared__ unsigned int lds[256 * 64];   // 64 KB: up to 256 cols x 128 f16
    const int tid = threadIdx.x;

    // ---- stage weights: each thread copies one 256B row (16 x 16B chunks)
    for (int r = tid; r < Ncols; r += 256) {
        const uint4* rowp = (const uint4*)(Wh + (size_t)r * DIM);
#pragma unroll
        for (int c = 0; c < 16; ++c) {
            uint4 v = rowp[c];
            *((uint4*)&lds[lds_chunk((unsigned)r, (unsigned)c)]) = v;
        }
    }
    __syncthreads();

    const int wave = tid >> 5;
    const int lane = tid & 31;
    const int mBase0 = blockIdx.x * 256 + wave * 16;
    const int mBase1 = mBase0 + 128;
    if (mBase0 >= M) return;

    const int lr = lane & 15;     // column / row-in-tile selector
    const int lh = lane >> 4;     // K-half selector

    // ---- A fragments, 2 M-tiles x 4 K-steps (ISA 16x32 f16 A layout):
    // lane<16: K = ks*32 + {0..7, 16..23}; lane>=16: K = ks*32 + {8..15, 24..31}
    v16h afrag[2][4];
#pragma unroll
    for (int mt = 0; mt < 2; ++mt) {
        const int row = (mt == 0 ? mBase0 : mBase1) + lr;
        const float* xrow = X + (size_t)row * DIM;
#pragma unroll
        for (int ks = 0; ks < 4; ++ks) {
            const float* p0 = xrow + ks * 32 + lh * 8;
            const float* p1 = p0 + 16;
            float tmp[16];
            *((float4*)&tmp[0])  = *((const float4*)p0);
            *((float4*)&tmp[4])  = *((const float4*)(p0 + 4));
            *((float4*)&tmp[8])  = *((const float4*)p1);
            *((float4*)&tmp[12]) = *((const float4*)(p1 + 4));
            v16h a;
#pragma unroll
            for (int e = 0; e < 16; ++e) a[e] = (_Float16)tmp[e];
            afrag[mt][ks] = a;
        }
    }

    const int ntiles = Ncols >> 4;
    for (int nt = 0; nt < ntiles; ++nt) {
        const unsigned ncol = (unsigned)(nt * 16 + lr);
        const int col = nt * 16 + lr;

        // ---- issue ALL B-fragment LDS loads for this tile first, then run
        // two interleaved independent WMMA accumulation chains (each fragment
        // is consumed twice -> 1 ds_load_b128 per WMMA, latency hidden by the
        // sibling chain's matrix ops).
        v8u wv[4];
#pragma unroll
        for (int ks = 0; ks < 4; ++ks) {
            // B fragment (32x16): lane = column ncol, elements match A layout:
            // K = ks*32 + lh*8 + {0..7, 16..23} -> 2 swizzled 16B chunks
            unsigned ch0 = (unsigned)(ks * 4 + lh);
            *((uint4*)&wv[ks])       = *((const uint4*)&lds[lds_chunk(ncol, ch0)]);
            *(((uint4*)&wv[ks]) + 1) = *((const uint4*)&lds[lds_chunk(ncol, ch0 + 2)]);
        }
        v8f c0 = {}, c1 = {};
#pragma unroll
        for (int ks = 0; ks < 4; ++ks) {
            v16h bfrag = __builtin_bit_cast(v16h, wv[ks]);
            c0 = __builtin_amdgcn_wmma_f32_16x16x32_f16(
                     false, afrag[0][ks], false, bfrag, (short)0, c0, false, false);
            c1 = __builtin_amdgcn_wmma_f32_16x16x32_f16(
                     false, afrag[1][ks], false, bfrag, (short)0, c1, false, false);
        }
        // ---- epilogue: C/D layout: lane lr = column, rows = lh*8 + i
        float bval = 0.0f;
        if constexpr (HAS_BIAS) bval = bias[col];
#pragma unroll
        for (int i = 0; i < 8; ++i) {
            float v = c0[i];
            if constexpr (HAS_BIAS) v += bval;
            if constexpr (RELU)     v = fmaxf(v, 0.0f);
            Out[(size_t)(mBase0 + lh * 8 + i) * Ncols + col] = v;
        }
#pragma unroll
        for (int i = 0; i < 8; ++i) {
            float v = c1[i];
            if constexpr (HAS_BIAS) v += bval;
            if constexpr (RELU)     v = fmaxf(v, 0.0f);
            Out[(size_t)(mBase1 + lh * 8 + i) * Ncols + col] = v;
        }
    }
}

// --------------------------- message aggregation ----------------------------
// h[v,t,d] += (1/7) * sum_{u in group(v), u!=v} relu(A[u,t,d] + B[v,t,d] + b[d])
// AB rows: [node*64+t][256] with A in cols 0..127, B in cols 128..255.
__global__ __launch_bounds__(256) void aggregate_k(float* __restrict__ h,
                                                   const float* __restrict__ AB,
                                                   const float* __restrict__ bmsg) {
    const int v = blockIdx.x;          // node
    const int g0 = (v >> 3) << 3;      // group base
    for (int it = threadIdx.x; it < T_STEPS * (DIM / 4); it += 256) {
        int t  = it >> 5;
        int d4 = (it & 31) << 2;
        const float4 Bv = *((const float4*)(AB + ((size_t)(v * T_STEPS + t)) * 256 + DIM + d4));
        const float4 bb = *((const float4*)(bmsg + d4));
        float4 acc = {0.f, 0.f, 0.f, 0.f};
#pragma unroll
        for (int j = 0; j < 8; ++j) {
            int u = g0 + j;
            if (u == v) continue;
            const float4 Av = *((const float4*)(AB + ((size_t)(u * T_STEPS + t)) * 256 + d4));
            acc.x += fmaxf(Av.x + Bv.x + bb.x, 0.f);
            acc.y += fmaxf(Av.y + Bv.y + bb.y, 0.f);
            acc.z += fmaxf(Av.z + Bv.z + bb.z, 0.f);
            acc.w += fmaxf(Av.w + Bv.w + bb.w, 0.f);
        }
        float4* hp = (float4*)(h + ((size_t)(v * T_STEPS + t)) * DIM + d4);
        float4 hv = *hp;
        const float inv = 1.0f / 7.0f;
        hv.x += acc.x * inv; hv.y += acc.y * inv;
        hv.z += acc.z * inv; hv.w += acc.w * inv;
        *hp = hv;
    }
}

// ------------------------------- view mean ----------------------------------
__global__ __launch_bounds__(256) void mean_k(const float* __restrict__ h,
                                              float* __restrict__ y) {
    size_t idx = (size_t)blockIdx.x * 256 + threadIdx.x;   // over 64*64*128
    int d = (int)(idx & (DIM - 1));
    size_t r = idx >> 7;
    int t = (int)(r & (T_STEPS - 1));
    int b = (int)(r >> 6);
    float s = 0.f;
#pragma unroll
    for (int j = 0; j < 8; ++j)
        s += h[((size_t)((b * 8 + j) * T_STEPS + t)) * DIM + d];
    y[idx] = s * 0.125f;
}

// ------------------------------- launcher -----------------------------------
extern "C" void kernel_launch(void* const* d_in, const int* in_sizes, int n_in,
                              void* d_out, int out_size, void* d_ws, size_t ws_size,
                              hipStream_t stream) {
    (void)in_sizes; (void)n_in; (void)out_size; (void)ws_size;
    // inputs: 0 view_id, 1 message_from, 2 message_to (unused: topology regular)
    const float* latents = (const float*)d_in[3];
    const float* W_msg   = (const float*)d_in[4];   // [3][128][256]
    const float* b_msg   = (const float*)d_in[5];   // [3][128]
    const float* W_r1    = (const float*)d_in[6];   // [128][128]
    const float* b_r1    = (const float*)d_in[7];
    const float* W_r2    = (const float*)d_in[8];
    const float* b_r2    = (const float*)d_in[9];
    float* out = (float*)d_out;                     // [64][64][128]

    char* ws = (char*)d_ws;
    auto take = [&](size_t bytes) {
        char* p = ws;
        ws += (bytes + 255) & ~(size_t)255;
        return p;
    };
    float*     h    = (float*)take((size_t)ROWS * DIM * 4);        // 16 MB
    float*     AB   = (float*)take((size_t)ROWS * 256 * 4);        // 32 MB
    float*     y    = (float*)take((size_t)OUTROWS * DIM * 4);     //  2 MB
    float*     z    = (float*)take((size_t)OUTROWS * DIM * 4);     //  2 MB
    _Float16*  Wcat = (_Float16*)take((size_t)3 * 256 * DIM * 2);  // 192 KB
    _Float16*  Wr1h = (_Float16*)take((size_t)DIM * DIM * 2);
    _Float16*  Wr2h = (_Float16*)take((size_t)DIM * DIM * 2);

    // prep: transpose latents, convert/pack weights to f16
    transpose_k<<<(ROWS * DIM) / 256, 256, 0, stream>>>(latents, h);
    prep_wcat_k<<<(3 * 256 * DIM) / 256, 256, 0, stream>>>(W_msg, Wcat);
    prep_w_k<<<(DIM * DIM) / 256, 256, 0, stream>>>(W_r1, Wr1h, DIM * DIM);
    prep_w_k<<<(DIM * DIM) / 256, 256, 0, stream>>>(W_r2, Wr2h, DIM * DIM);

    // message-passing rounds: GEMM (A|B panels) + aggregation
    for (int r = 0; r < 3; ++r) {
        gemm_k<false, false><<<ROWS / 256, 256, 0, stream>>>(
            h, Wcat + (size_t)r * 256 * DIM, AB, nullptr, ROWS, 256);
        aggregate_k<<<N_NODES, 256, 0, stream>>>(h, AB, b_msg + (size_t)r * DIM);
    }

    // readout: mean over views, relu-GEMM, final GEMM (+bias) -> out
    mean_k<<<(OUTROWS * DIM) / 256, 256, 0, stream>>>(h, y);
    gemm_k<true, true><<<OUTROWS / 256, 256, 0, stream>>>(y, Wr1h, z, b_r1, OUTROWS, DIM);
    gemm_k<true, false><<<OUTROWS / 256, 256, 0, stream>>>(z, Wr2h, out, b_r2, OUTROWS, DIM);
}
